// _BaseMultiHeadAttention_30940944401079
// MI455X (gfx1250) — matrix operand order repaired and checked by reading, not compiled, not run on hardware
//
#include <hip/hip_runtime.h>

// ---------------------------------------------------------------------------
// Problem constants (match reference)
// ---------------------------------------------------------------------------
constexpr int B_  = 8;
constexpr int S_  = 2048;
constexpr int H_  = 16;
constexpr int D_  = 64;
constexpr int DM_ = H_ * D_;        // 1024
constexpr int W_  = 4;
constexpr int SC_ = S_ / W_;        // 512
constexpr int BH_ = B_ * H_;        // 128
constexpr int KF_ = W_ * D_;        // 256 conv reduction dim

typedef __attribute__((ext_vector_type(16))) __bf16 bf16x16;
typedef __attribute__((ext_vector_type(8)))  float  f32x8;

union FragU {
    bf16x16 v;
    uint4   q[2];
    __bf16  h[16];
};

__device__ __forceinline__ f32x8 wmma_bf16(bf16x16 a, bf16x16 b, f32x8 c) {
    return __builtin_amdgcn_wmma_f32_16x16x32_bf16(
        /*neg_a=*/false, a, /*neg_b=*/false, b,
        /*c_mod=*/(short)0, c, /*reuse_a=*/false, /*reuse_b=*/false);
}

// A fragment (16x32, bf16) from row-major [rows x ld] matrix.
// lane<16: row = row0+lane,    halves = K{k0..k0+7}  and K{k0+16..k0+23}
// lane>=16: row = row0+lane-16, halves = K{k0+8..k0+15} and K{k0+24..k0+31}
__device__ __forceinline__ bf16x16 load_a_frag(const __bf16* M, int ld,
                                               int row0, int k0, int lane) {
    FragU f;
    int r    = row0 + (lane & 15);
    int kofs = k0 + ((lane & 16) ? 8 : 0);
    f.q[0] = *(const uint4*)(M + (size_t)r * ld + kofs);
    f.q[1] = *(const uint4*)(M + (size_t)r * ld + kofs + 16);
    return f.v;
}

// B fragment (32x16, bf16) from an N-major store Bn[N][K] with K-stride ld.
// lane<16: col = n0+lane, K = k0..k0+7, k0+16..k0+23 ; lane>=16: col = n0+lane-16, K = k0+8..k0+15, k0+24..k0+31
__device__ __forceinline__ FragU load_b_raw(const __bf16* Bn, int ld,
                                            int k0, int n0, int lane) {
    FragU f;
    int n    = n0 + (lane & 15);
    int kofs = k0 + ((lane & 16) ? 8 : 0);
    f.q[0] = *(const uint4*)(Bn + (size_t)n * ld + kofs);
    f.q[1] = *(const uint4*)(Bn + (size_t)n * ld + kofs + 16);
    return f;
}

// Reductions across the 16 lanes of each wave-half (xor masks stay inside half)
__device__ __forceinline__ float half_max(float v) {
    #pragma unroll
    for (int m = 1; m < 16; m <<= 1) v = fmaxf(v, __shfl_xor(v, m, 32));
    return v;
}
__device__ __forceinline__ float half_sum(float v) {
    #pragma unroll
    for (int m = 1; m < 16; m <<= 1) v += __shfl_xor(v, m, 32);
    return v;
}

// ---------------------------------------------------------------------------
// Prep kernels: layout + bf16 conversion (native __bf16 truncation)
// ---------------------------------------------------------------------------
__global__ void prep_q_kernel(const float* __restrict__ in, __bf16* __restrict__ out, int n) {
    int i = blockIdx.x * blockDim.x + threadIdx.x;
    if (i >= n) return;
    int d = i & 63;
    int h = (i >> 6) & 15;
    int s = (i >> 10) & 2047;
    int b = i >> 21;
    out[(((size_t)b * H_ + h) * S_ + s) * D_ + d] = (__bf16)in[i];
}

__global__ void prep_wT_kernel(const float* __restrict__ w, __bf16* __restrict__ wT) {
    int i = blockIdx.x * blockDim.x + threadIdx.x;
    if (i >= DM_ * DM_) return;
    int k = i >> 10, n = i & 1023;            // w[k][n] -> wT[n][k]
    wT[(size_t)n * DM_ + k] = (__bf16)w[i];
}

// conv kernel [W][D][D] = [w][din][dout]  ->  wT[dout][w*64+din]
__global__ void prep_convT_kernel(const float* __restrict__ kk, __bf16* __restrict__ kT) {
    int i = blockIdx.x * blockDim.x + threadIdx.x;
    if (i >= D_ * KF_) return;
    int dout = i >> 8, f = i & 255;
    int w = f >> 6, din = f & 63;
    kT[i] = (__bf16)kk[((size_t)w * D_ + din) * D_ + dout];
}

// ---------------------------------------------------------------------------
// Compression: per (b,h, 16-t tile)  [16 x 256] x [256 x 64] bf16 WMMA GEMM
// transposed==0 -> out = kc [BH][Sc][64]; transposed==1 -> out = vcT [BH][64][Sc]
// ---------------------------------------------------------------------------
__global__ __launch_bounds__(32) void compress_kernel(
    const float* __restrict__ pre,           // [B,S,H,D] fp32
    const __bf16* __restrict__ wT,           // [64][256] bf16
    const float* __restrict__ bias,          // [64]
    __bf16* __restrict__ out,
    int transposed)
{
    int lane = threadIdx.x;
    int bh = blockIdx.x;
    int b = bh >> 4, h = bh & 15;
    int t0 = blockIdx.y * 16;
    const float* base = pre + (size_t)b * S_ * H_ * D_ + (size_t)h * D_;

    f32x8 acc[4] = {{}, {}, {}, {}};

    #pragma unroll
    for (int c = 0; c < KF_ / 32; ++c) {
        int k0 = c * 32;
        // Build A fragment, converting fp32 windows to bf16 on the fly.
        FragU fa;
        int t    = t0 + (lane & 15);
        int kofs = k0 + ((lane & 16) ? 8 : 0);
        #pragma unroll
        for (int g = 0; g < 2; ++g) {
            int f0  = kofs + g * 16;           // 8 consecutive features, same w
            int w   = f0 >> 6;
            int din = f0 & 63;
            const float* src = base + (size_t)(t * W_ + w) * (H_ * D_) + din;
            float4 x0 = *(const float4*)src;
            float4 x1 = *(const float4*)(src + 4);
            fa.h[g * 8 + 0] = (__bf16)x0.x; fa.h[g * 8 + 1] = (__bf16)x0.y;
            fa.h[g * 8 + 2] = (__bf16)x0.z; fa.h[g * 8 + 3] = (__bf16)x0.w;
            fa.h[g * 8 + 4] = (__bf16)x1.x; fa.h[g * 8 + 5] = (__bf16)x1.y;
            fa.h[g * 8 + 6] = (__bf16)x1.z; fa.h[g * 8 + 7] = (__bf16)x1.w;
        }
        bf16x16 a = fa.v;
        #pragma unroll
        for (int nt = 0; nt < 4; ++nt) {
            FragU bb = load_b_raw(wT, KF_, k0, nt * 16, lane);
            acc[nt] = wmma_bf16(a, bb.v, acc[nt]);
        }
    }

    int mb = (lane & 16) ? 8 : 0;
    int nn = lane & 15;
    #pragma unroll
    for (int nt = 0; nt < 4; ++nt) {
        int dout = nt * 16 + nn;
        float bv = bias[dout];
        #pragma unroll
        for (int r = 0; r < 8; ++r) {
            int t = t0 + mb + r;
            __bf16 val = (__bf16)(acc[nt][r] + bv);
            if (transposed)
                out[(size_t)bh * D_ * SC_ + (size_t)dout * SC_ + t] = val;
            else
                out[(size_t)bh * SC_ * D_ + (size_t)t * D_ + dout] = val;
        }
    }
}

// ---------------------------------------------------------------------------
// Attention: one wave per (b,h, 16 queries). Flash-style online softmax over
// Sc=512 in 32-column steps. Scores & PV via bf16 WMMA. kc/vc fragments are
// double-buffered so next-tile global loads overlap the softmax VALU chain;
// global_prefetch pulls the kt+2 region toward the WGP.
// ---------------------------------------------------------------------------
__global__ __launch_bounds__(32) void attn_kernel(
    const __bf16* __restrict__ qbf,   // [B,H,S,D] bf16
    const __bf16* __restrict__ kc,    // [BH][Sc][D] bf16
    const __bf16* __restrict__ vcT,   // [BH][D][Sc] bf16
    __bf16* __restrict__ attn_out)    // [B,S,DM] bf16
{
    __shared__ __bf16 pl[16 * 32];    // P tile staged for A-fragment reload
    int lane = threadIdx.x;
    int bh = blockIdx.x;
    int b = bh >> 4, h = bh & 15;
    int s0 = blockIdx.y * 16;

    const __bf16* qp  = qbf + ((size_t)bh * S_ + s0) * D_;
    const __bf16* kcp = kc  + (size_t)bh * SC_ * D_;
    const __bf16* vtp = vcT + (size_t)bh * D_ * SC_;

    bf16x16 aq0 = load_a_frag(qp, D_, 0, 0,  lane);
    bf16x16 aq1 = load_a_frag(qp, D_, 0, 32, lane);

    f32x8 acc0 = {}, acc1 = {}, acc2 = {}, acc3 = {};
    float rmax[8], rsum[8];
    #pragma unroll
    for (int r = 0; r < 8; ++r) { rmax[r] = -1e30f; rsum[r] = 0.f; }
    const float scale = 0.125f;               // 1/sqrt(64)
    constexpr int NKT = SC_ / 32;             // 16 iterations

    // ---- prologue: load fragments for tile 0 ----
    FragU kb0 = load_b_raw(kcp, D_, 0,  0,  lane);
    FragU kb1 = load_b_raw(kcp, D_, 32, 0,  lane);
    FragU kb2 = load_b_raw(kcp, D_, 0,  16, lane);
    FragU kb3 = load_b_raw(kcp, D_, 32, 16, lane);
    FragU vb0 = load_b_raw(vtp, SC_, 0, 0,  lane);
    FragU vb1 = load_b_raw(vtp, SC_, 0, 16, lane);
    FragU vb2 = load_b_raw(vtp, SC_, 0, 32, lane);
    FragU vb3 = load_b_raw(vtp, SC_, 0, 48, lane);

    for (int kt = 0; kt < NKT; ++kt) {
        int t0 = kt * 32;
        int tn = (kt + 1 < NKT) ? t0 + 32 : 0;     // next tile (wraps harmlessly)

        // ---- scores for current tile (consume kb*) ----
        f32x8 sA = {}, sB = {};
        sA = wmma_bf16(aq0, kb0.v, sA);
        sA = wmma_bf16(aq1, kb1.v, sA);
        sB = wmma_bf16(aq0, kb2.v, sB);
        sB = wmma_bf16(aq1, kb3.v, sB);

        // ---- issue next-tile loads; they overlap the softmax below ----
        FragU nk0 = load_b_raw(kcp, D_, 0,  tn,      lane);
        FragU nk1 = load_b_raw(kcp, D_, 32, tn,      lane);
        FragU nk2 = load_b_raw(kcp, D_, 0,  tn + 16, lane);
        FragU nk3 = load_b_raw(kcp, D_, 32, tn + 16, lane);
        FragU nv0 = load_b_raw(vtp, SC_, tn, 0,  lane);
        FragU nv1 = load_b_raw(vtp, SC_, tn, 16, lane);
        FragU nv2 = load_b_raw(vtp, SC_, tn, 32, lane);
        FragU nv3 = load_b_raw(vtp, SC_, tn, 48, lane);

        // L2->WGP prefetch for the tile after next
        if (kt + 2 < NKT) {
            __builtin_prefetch(kcp + (size_t)(t0 + 64 + (lane & 15)) * D_, 0, 1);
            __builtin_prefetch(vtp + (size_t)lane * SC_ + t0 + 64, 0, 1);
            __builtin_prefetch(vtp + (size_t)(lane + 32) * SC_ + t0 + 64, 0, 1);
        }

        // ---- online softmax update ----
        float alpha[8];
        #pragma unroll
        for (int r = 0; r < 8; ++r) {
            float a = sA[r] * scale, c = sB[r] * scale;
            float tmax = half_max(fmaxf(a, c));
            float nm = fmaxf(rmax[r], tmax);
            alpha[r] = __expf(rmax[r] - nm);
            rmax[r] = nm;
            float pa = __expf(a - nm), pb = __expf(c - nm);
            sA[r] = pa; sB[r] = pb;
            rsum[r] = rsum[r] * alpha[r] + half_sum(pa + pb);
        }

        // ---- stage P (16x32 bf16) to LDS for A-fragment layout reload ----
        int mb = (lane & 16) ? 8 : 0;
        int nn = lane & 15;
        #pragma unroll
        for (int r = 0; r < 8; ++r) {
            pl[(mb + r) * 32 + nn]      = (__bf16)sA[r];
            pl[(mb + r) * 32 + 16 + nn] = (__bf16)sB[r];
        }
        __syncthreads();
        bf16x16 pf = load_a_frag(pl, 32, 0, 0, lane);

        // ---- rescale accumulators, then accumulate P x V (K=32) ----
        #pragma unroll
        for (int r = 0; r < 8; ++r) {
            acc0[r] *= alpha[r]; acc1[r] *= alpha[r];
            acc2[r] *= alpha[r]; acc3[r] *= alpha[r];
        }
        acc0 = wmma_bf16(pf, vb0.v, acc0);
        acc1 = wmma_bf16(pf, vb1.v, acc1);
        acc2 = wmma_bf16(pf, vb2.v, acc2);
        acc3 = wmma_bf16(pf, vb3.v, acc3);
        __syncthreads();

        // ---- rotate double buffer ----
        kb0 = nk0; kb1 = nk1; kb2 = nk2; kb3 = nk3;
        vb0 = nv0; vb1 = nv1; vb2 = nv2; vb3 = nv3;
    }

    // ---- epilogue: normalize, store to [B,S,DM] bf16 ----
    int mb = (lane & 16) ? 8 : 0;
    int nn = lane & 15;
    #pragma unroll
    for (int r = 0; r < 8; ++r) {
        float inv = 1.0f / rsum[r];
        size_t row = ((size_t)b * S_ + s0 + mb + r) * DM_ + (size_t)h * D_;
        attn_out[row + nn]      = (__bf16)(acc0[r] * inv);
        attn_out[row + 16 + nn] = (__bf16)(acc1[r] * inv);
        attn_out[row + 32 + nn] = (__bf16)(acc2[r] * inv);
        attn_out[row + 48 + nn] = (__bf16)(acc3[r] * inv);
    }
}

// ---------------------------------------------------------------------------
// Output projection: Y[16384,1024] = X[16384,1024] x W[1024,1024], fp32 out.
// One wave per 32x64 output tile (2 A-frags amortize each B-frag set).
// ---------------------------------------------------------------------------
__global__ __launch_bounds__(32) void proj_kernel(
    const __bf16* __restrict__ X,    // [B*S, DM] bf16
    const __bf16* __restrict__ WT,   // [DM][DM] bf16 (N-major)
    float* __restrict__ Y)           // [B*S, DM] fp32
{
    int lane = threadIdx.x;
    int m0 = blockIdx.x * 32;
    int n0 = blockIdx.y * 64;

    f32x8 acc[2][4] = {{{}, {}, {}, {}}, {{}, {}, {}, {}}};
    for (int c = 0; c < DM_ / 32; ++c) {
        int k0 = c * 32;
        bf16x16 a0 = load_a_frag(X, DM_, m0,      k0, lane);
        bf16x16 a1 = load_a_frag(X, DM_, m0 + 16, k0, lane);
        #pragma unroll
        for (int nt = 0; nt < 4; ++nt) {
            FragU bb = load_b_raw(WT, DM_, k0, n0 + nt * 16, lane);
            acc[0][nt] = wmma_bf16(a0, bb.v, acc[0][nt]);
            acc[1][nt] = wmma_bf16(a1, bb.v, acc[1][nt]);
        }
    }

    int mb = (lane & 16) ? 8 : 0;
    int nn = lane & 15;
    #pragma unroll
    for (int half = 0; half < 2; ++half)
        #pragma unroll
        for (int nt = 0; nt < 4; ++nt)
            #pragma unroll
            for (int r = 0; r < 8; ++r)
                Y[(size_t)(m0 + half * 16 + mb + r) * DM_ + n0 + nt * 16 + nn] = acc[half][nt][r];
}

// ---------------------------------------------------------------------------
// Host launcher
// ---------------------------------------------------------------------------
extern "C" void kernel_launch(void* const* d_in, const int* in_sizes, int n_in,
                              void* d_out, int out_size, void* d_ws, size_t ws_size,
                              hipStream_t stream) {
    (void)in_sizes; (void)n_in; (void)out_size; (void)ws_size;

    const float* pre_q  = (const float*)d_in[0];
    const float* pre_k  = (const float*)d_in[1];
    const float* pre_v  = (const float*)d_in[2];
    const float* k_conv = (const float*)d_in[3];
    const float* k_bias = (const float*)d_in[4];
    const float* v_conv = (const float*)d_in[5];
    const float* v_bias = (const float*)d_in[6];
    const float* out_w  = (const float*)d_in[7];
    float* out = (float*)d_out;

    char* ws = (char*)d_ws;
    const size_t QB   = (size_t)B_ * H_ * S_ * D_ * 2;       // 32 MB
    const size_t AT   = (size_t)B_ * S_ * DM_ * 2;           // 32 MB
    const size_t KC   = (size_t)BH_ * SC_ * D_ * 2;          // 8 MB
    __bf16* qbf  = (__bf16*)(ws);
    __bf16* attn = (__bf16*)(ws + QB);
    __bf16* kc   = (__bf16*)(ws + QB + AT);
    __bf16* vcT  = (__bf16*)(ws + QB + AT + KC);
    __bf16* WT   = (__bf16*)(ws + QB + AT + 2 * KC);
    __bf16* kwT  = (__bf16*)(ws + QB + AT + 2 * KC + (size_t)DM_ * DM_ * 2);
    __bf16* vwT  = kwT + D_ * KF_;

    const int NQ = B_ * S_ * H_ * D_;
    prep_q_kernel<<<(NQ + 255) / 256, 256, 0, stream>>>(pre_q, qbf, NQ);
    prep_wT_kernel<<<(DM_ * DM_ + 255) / 256, 256, 0, stream>>>(out_w, WT);
    prep_convT_kernel<<<(D_ * KF_ + 255) / 256, 256, 0, stream>>>(k_conv, kwT);
    prep_convT_kernel<<<(D_ * KF_ + 255) / 256, 256, 0, stream>>>(v_conv, vwT);

    compress_kernel<<<dim3(BH_, SC_ / 16), 32, 0, stream>>>(pre_k, kwT, k_bias, kc, 0);
    compress_kernel<<<dim3(BH_, SC_ / 16), 32, 0, stream>>>(pre_v, vwT, v_bias, vcT, 1);

    attn_kernel<<<dim3(BH_, S_ / 16), 32, 0, stream>>>(qbf, kc, vcT, attn);

    proj_kernel<<<dim3((B_ * S_) / 32, DM_ / 64), 32, 0, stream>>>(attn, WT, out);
}
